// FeaturePropagation_24524263260776
// MI455X (gfx1250) — matrix operand order repaired and checked by reading, not compiled, not run on hardware
//
#include <hip/hip_runtime.h>
#include <hip/hip_bf16.h>

typedef _Float16 half_t;
typedef __attribute__((ext_vector_type(16))) _Float16 v16h;
typedef __attribute__((ext_vector_type(8)))  _Float16 v8h;
typedef __attribute__((ext_vector_type(8)))  float    v8f;

// Problem constants (from setup_inputs)
#define BB   4
#define NN   16384
#define SS   4096
#define C1   128
#define C2   256
#define CIN  384
#define H1   256
#define H2   128
#define MM   (BB * NN)   // 65536 rows

// ---------------- workspace layout (bytes, all 2KB aligned) ----------------
#define OFF_XF16   ((size_t)0)                          // M*CIN f16   = 50331648
#define OFF_W1H    ((size_t)50331648)                   // H1*CIN f16  = 196608
#define OFF_W2H    ((size_t)50528256)                   // H2*H1  f16  = 65536
#define OFF_Y1     ((size_t)50593792)                   // M*H1 f32    = 67108864
#define OFF_H1F16  ((size_t)117702656)                  // M*H1 f16    = 33554432
#define OFF_PART1  ((size_t)151257088)                  // 64*2*H1 f32 = 131072
#define OFF_PART2  ((size_t)151388160)                  // 64*2*H2 f32 = 65536
#define OFF_STAT1  ((size_t)151453696)                  // 2*H1 f32
#define OFF_STAT2  ((size_t)151455744)                  // 2*H2 f32

// ---------------------------------------------------------------------------
// Convert weights fp32 -> fp16 once per call
__global__ void prep_weights(const float* __restrict__ W1, const float* __restrict__ W2,
                             half_t* __restrict__ W1h, half_t* __restrict__ W2h) {
    int i = blockIdx.x * blockDim.x + threadIdx.x;
    if (i < H1 * CIN) W1h[i] = (half_t)W1[i];
    if (i < H2 * H1)  W2h[i] = (half_t)W2[i];
}

// ---------------------------------------------------------------------------
// 3-NN inverse-distance interpolation + concat -> X f16 (M x 384)
// One block = 256 query points of one batch; xyz2[b] staged SoA in LDS (48KB).
__global__ __launch_bounds__(256) void interp_kernel(
    const float* __restrict__ xyz1, const float* __restrict__ xyz2,
    const float* __restrict__ f1,   const float* __restrict__ f2,
    half_t* __restrict__ X) {
    __shared__ float lx[SS], ly[SS], lz[SS];
    const int b = blockIdx.y;
    const float* x2 = xyz2 + (size_t)b * SS * 3;
    for (int i = threadIdx.x; i < SS * 3; i += blockDim.x) {
        float v = x2[i];
        int s = i / 3, c = i - s * 3;
        if (c == 0) lx[s] = v; else if (c == 1) ly[s] = v; else lz[s] = v;
    }
    __syncthreads();

    const int n = blockIdx.x * blockDim.x + threadIdx.x;      // 0..NN-1
    const float* p = xyz1 + ((size_t)b * NN + n) * 3;
    const float px = p[0], py = p[1], pz = p[2];

    float d0 = 1e30f, d1 = 1e30f, d2 = 1e30f;
    int   i0 = 0,     i1 = 0,     i2 = 0;
    #pragma unroll 4
    for (int s = 0; s < SS; ++s) {
        float dx = px - lx[s], dy = py - ly[s], dz = pz - lz[s];
        float d = dx * dx + dy * dy + dz * dz;
        if (d < d2) {
            if (d < d1) {
                d2 = d1; i2 = i1;
                if (d < d0) { d1 = d0; i1 = i0; d0 = d; i0 = s; }
                else        { d1 = d;  i1 = s; }
            } else { d2 = d; i2 = s; }
        }
    }
    float w0 = 1.f / (d0 + 1e-8f);
    float w1 = 1.f / (d1 + 1e-8f);
    float w2 = 1.f / (d2 + 1e-8f);
    float ws = w0 + w1 + w2;
    w0 /= ws; w1 /= ws; w2 /= ws;

    const float4* r0 = (const float4*)(f2 + ((size_t)b * SS + i0) * C2);
    const float4* r1 = (const float4*)(f2 + ((size_t)b * SS + i1) * C2);
    const float4* r2 = (const float4*)(f2 + ((size_t)b * SS + i2) * C2);
    half_t* xr = X + ((size_t)b * NN + n) * CIN;
    #pragma unroll 4
    for (int c = 0; c < C2 / 4; ++c) {
        float4 a = r0[c], bv = r1[c], cv = r2[c];
        xr[4 * c + 0] = (half_t)(w0 * a.x + w1 * bv.x + w2 * cv.x);
        xr[4 * c + 1] = (half_t)(w0 * a.y + w1 * bv.y + w2 * cv.y);
        xr[4 * c + 2] = (half_t)(w0 * a.z + w1 * bv.z + w2 * cv.z);
        xr[4 * c + 3] = (half_t)(w0 * a.w + w1 * bv.w + w2 * cv.w);
    }
    const float4* fr = (const float4*)(f1 + ((size_t)b * NN + n) * C1);
    #pragma unroll 4
    for (int c = 0; c < C1 / 4; ++c) {
        float4 a = fr[c];
        xr[C2 + 4 * c + 0] = (half_t)a.x;
        xr[C2 + 4 * c + 1] = (half_t)a.y;
        xr[C2 + 4 * c + 2] = (half_t)a.z;
        xr[C2 + 4 * c + 3] = (half_t)a.w;
    }
}

// ---------------------------------------------------------------------------
// WMMA fragment loaders (layouts per CDNA5 ISA 7.12.2, 16-bit 16x32 A / 32x16 B)
__device__ inline v16h load_a_frag(const half_t* __restrict__ A, int K, int m0,
                                   int k0, int lane) {
    const half_t* row = A + (size_t)(m0 + (lane & 15)) * K + k0 + ((lane >> 4) << 3);
    v8h lo = *(const v8h*)(row);
    v8h hi = *(const v8h*)(row + 16);
    v16h r;
    #pragma unroll
    for (int i = 0; i < 8; ++i) { r[i] = lo[i]; r[8 + i] = hi[i]; }
    return r;
}

__device__ inline v16h load_b_frag(const half_t* __restrict__ W, int K, int n0,
                                   int k0, int lane) {
    // W is row-major (N x K) == B^T; lane owns one column of B (row of W)
    const half_t* row = W + (size_t)(n0 + (lane & 15)) * K + k0 + ((lane >> 4) << 3);
    v8h lo = *(const v8h*)(row);
    v8h hi = *(const v8h*)(row + 16);
    v16h r;
    #pragma unroll
    for (int i = 0; i < 8; ++i) { r[i] = lo[i]; r[8 + i] = hi[i]; }
    return r;
}

__device__ inline void store_tile(float* __restrict__ Y, int N,
                                  const float* __restrict__ bias,
                                  v8f acc, int m0, int n0, int lane) {
    const int n  = n0 + (lane & 15);
    const float bv = bias[n];
    const int mb = m0 + ((lane >> 4) << 3);
    #pragma unroll
    for (int v = 0; v < 8; ++v)
        Y[(size_t)(mb + v) * N + n] = acc[v] + bv;
}

// Y[M x N] = A[M x K] * W^T + bias
// 2x2 register tiling: each block owns 32 rows (2 m-tiles); wave w owns
// n-tile w (and n-tile w+8 when TWO_N). TWO_N is a compile-time parameter so
// the inner loop is branch-free straight-line code: 4 frag loads -> 4 WMMAs.
template <bool TWO_N>
__global__ __launch_bounds__(256) void gemm_wmma(
    const half_t* __restrict__ A, const half_t* __restrict__ W,
    const float* __restrict__ bias, float* __restrict__ Y,
    int M, int N, int K) {
    const int lane = threadIdx.x & 31;
    const int wave = threadIdx.x >> 5;
    const int m0 = blockIdx.x << 5;                 // 32 rows per block
    const int n0 = wave << 4;
    const int n1 = (wave + 8) << 4;

    v8f acc00 = {}, acc10 = {}, acc01 = {}, acc11 = {};
    for (int k0 = 0; k0 < K; k0 += 32) {
        v16h a0 = load_a_frag(A, K, m0,      k0, lane);
        v16h a1 = load_a_frag(A, K, m0 + 16, k0, lane);
        v16h b0 = load_b_frag(W, K, n0, k0, lane);
        acc00 = __builtin_amdgcn_wmma_f32_16x16x32_f16(
            false, a0, false, b0, (short)0, acc00, false, false);
        acc10 = __builtin_amdgcn_wmma_f32_16x16x32_f16(
            false, a1, false, b0, (short)0, acc10, false, false);
        if (TWO_N) {
            v16h b1 = load_b_frag(W, K, n1, k0, lane);
            acc01 = __builtin_amdgcn_wmma_f32_16x16x32_f16(
                false, a0, false, b1, (short)0, acc01, false, false);
            acc11 = __builtin_amdgcn_wmma_f32_16x16x32_f16(
                false, a1, false, b1, (short)0, acc11, false, false);
        }
    }
    store_tile(Y, N, bias, acc00, m0,      n0, lane);
    store_tile(Y, N, bias, acc10, m0 + 16, n0, lane);
    if (TWO_N) {
        store_tile(Y, N, bias, acc01, m0,      n1, lane);
        store_tile(Y, N, bias, acc11, m0 + 16, n1, lane);
    }
}

// ---------------------------------------------------------------------------
// Deterministic BN stats: phase 1 per-block partial sums (blockDim.x == H)
__global__ void stats_partial(const float* __restrict__ Y, float* __restrict__ part,
                              int M, int H) {
    const int c = threadIdx.x;
    const int rows = M / gridDim.x;
    const int r0 = blockIdx.x * rows;
    float s = 0.f, ss = 0.f;
    for (int r = r0; r < r0 + rows; ++r) {
        float v = Y[(size_t)r * H + c];
        s += v; ss += v * v;
    }
    part[(size_t)blockIdx.x * 2 * H + c]     = s;
    part[(size_t)blockIdx.x * 2 * H + H + c] = ss;
}

// phase 2: fixed-order combine -> mean/var
__global__ void stats_final(const float* __restrict__ part, float* __restrict__ stats,
                            int nblocks, int H, int M) {
    const int c = threadIdx.x;
    if (c >= H) return;
    float s = 0.f, ss = 0.f;
    for (int i = 0; i < nblocks; ++i) {
        s  += part[(size_t)i * 2 * H + c];
        ss += part[(size_t)i * 2 * H + H + c];
    }
    float mean = s / (float)M;
    float var  = ss / (float)M - mean * mean;
    stats[c]     = mean;
    stats[H + c] = var;
}

// BN (training stats) + ReLU; F16OUT writes f16 activations for the next GEMM
template <bool F16OUT>
__global__ void bn_relu(const float* __restrict__ Y, const float* __restrict__ stats,
                        const float* __restrict__ g, const float* __restrict__ be,
                        void* __restrict__ out, int M, int H) {
    size_t idx = (size_t)blockIdx.x * blockDim.x + threadIdx.x;
    const size_t total = (size_t)M * H;
    const size_t stride = (size_t)gridDim.x * blockDim.x;
    for (; idx < total; idx += stride) {
        const int c = (int)(idx & (H - 1));           // H is a power of two
        const float mean = stats[c], var = stats[H + c];
        const float sc = g[c] * rsqrtf(var + 1e-5f);
        float v = (Y[idx] - mean) * sc + be[c];
        v = fmaxf(v, 0.f);
        if (F16OUT) ((half_t*)out)[idx] = (half_t)v;
        else        ((float*)out)[idx]  = v;
    }
}

// ---------------------------------------------------------------------------
extern "C" void kernel_launch(void* const* d_in, const int* in_sizes, int n_in,
                              void* d_out, int out_size, void* d_ws, size_t ws_size,
                              hipStream_t stream) {
    (void)in_sizes; (void)n_in; (void)out_size; (void)ws_size;
    const float* xyz1 = (const float*)d_in[0];
    const float* xyz2 = (const float*)d_in[1];
    const float* f1   = (const float*)d_in[2];
    const float* f2   = (const float*)d_in[3];
    const float* W1   = (const float*)d_in[4];
    const float* b1   = (const float*)d_in[5];
    const float* g1   = (const float*)d_in[6];
    const float* be1  = (const float*)d_in[7];
    const float* W2   = (const float*)d_in[8];
    const float* b2   = (const float*)d_in[9];
    const float* g2   = (const float*)d_in[10];
    const float* be2  = (const float*)d_in[11];

    char* ws = (char*)d_ws;
    half_t* Xf16  = (half_t*)(ws + OFF_XF16);
    half_t* W1h   = (half_t*)(ws + OFF_W1H);
    half_t* W2h   = (half_t*)(ws + OFF_W2H);
    float*  Y1    = (float*)(ws + OFF_Y1);
    half_t* H1f16 = (half_t*)(ws + OFF_H1F16);
    float*  part1 = (float*)(ws + OFF_PART1);
    float*  part2 = (float*)(ws + OFF_PART2);
    float*  stat1 = (float*)(ws + OFF_STAT1);
    float*  stat2 = (float*)(ws + OFF_STAT2);
    float*  Y2    = (float*)d_out;                 // d_out doubles as layer-2 buffer

    // 1. fp32 -> fp16 weights
    prep_weights<<<(H1 * CIN + 255) / 256, 256, 0, stream>>>(W1, W2, W1h, W2h);

    // 2. 3-NN interpolate + concat -> X f16 (M x 384)
    interp_kernel<<<dim3(NN / 256, BB), 256, 0, stream>>>(xyz1, xyz2, f1, f2, Xf16);

    // 3. layer 1: Y1 = X * W1^T + b1   (WMMA f16 -> f32, 2x2 register tiles)
    gemm_wmma<true><<<MM / 32, 256, 0, stream>>>(Xf16, W1h, b1, Y1, MM, H1, CIN);

    // 4-5. deterministic BN stats for layer 1
    stats_partial<<<64, H1, 0, stream>>>(Y1, part1, MM, H1);
    stats_final<<<1, H1, 0, stream>>>(part1, stat1, 64, H1, MM);

    // 6. BN + ReLU -> f16 activations
    bn_relu<true><<<2048, 256, 0, stream>>>(Y1, stat1, g1, be1, H1f16, MM, H1);

    // 7. layer 2: Y2 = H1 * W2^T + b2 (into d_out, 1 n-tile per wave)
    gemm_wmma<false><<<MM / 32, 256, 0, stream>>>(H1f16, W2h, b2, Y2, MM, H2, H1);

    // 8-9. deterministic BN stats for layer 2
    stats_partial<<<64, H2, 0, stream>>>(Y2, part2, MM, H2);
    stats_final<<<1, H2, 0, stream>>>(part2, stat2, 64, H2, MM);

    // 10. BN + ReLU in place -> final f32 output
    bn_relu<false><<<2048, 256, 0, stream>>>(Y2, stat2, g2, be2, Y2, MM, H2);
}
